// SelectiveSSM_13640816132180
// MI455X (gfx1250) — compile-verified, matrix operand order checked by reading, not run on hardware
//
#include <hip/hip_runtime.h>
#include <cstdint>
#include <cstddef>

// ---------------- problem constants (from the reference) ----------------
#define DM    1024            // D_MODEL
#define DS    16              // D_STATE
#define DC    4               // D_CONV
#define DI    2048            // D_INNER
#define DTR   64              // DT_RANK
#define BSZ   4
#define LSEQ  4096
#define MTOT  (BSZ*LSEQ)      // 16384 rows for all GEMMs
#define XPW   (DTR + 2*DS)    // 96 = xproj output width
#define CHK   256             // scan chunk length
#define NCHK  (LSEQ / CHK)    // 16 chunks per sequence

// ---------------- WMMA types ----------------
typedef __attribute__((ext_vector_type(16))) __bf16 v16bf;
typedef __attribute__((ext_vector_type(8)))  __bf16 v8bf;
typedef __attribute__((ext_vector_type(8)))  float  v8f;
typedef __attribute__((ext_vector_type(4)))  int    v4i;

union Frag16 { v16bf v; v8bf h[2]; };
union AccU   { v8f v;   float f[8]; };

#define GLOBAL_AS __attribute__((address_space(1)))
#define LDS_AS    __attribute__((address_space(3)))

// ------- CDNA5 async global->LDS path (probe with __has_builtin) -------
#if defined(__has_builtin)
# if __has_builtin(__builtin_amdgcn_global_load_async_to_lds_b128)
#  define ASYNC_OK 1
# endif
#endif
#ifndef ASYNC_OK
# define ASYNC_OK 0
#endif

#if ASYNC_OK
# if __has_builtin(__builtin_amdgcn_s_wait_asynccnt)
#  define WAIT_ASYNC(n) __builtin_amdgcn_s_wait_asynccnt(n)
# else
#  define WAIT_ASYNC(n) asm volatile("s_wait_asynccnt %0" :: "n"(n))
# endif
#else
# define WAIT_ASYNC(n) ((void)0)
#endif

// Fragment layout (CDNA5 wave32, 16-bit operands): lane&15 = row (M for A /
// N for B); lane>>4 selects K-subchunk: regs 0-3 hold K=kc..kc+7, regs 4-7
// hold K=kc+16..kc+23, kc=(lane>>4)*8.  Two 16-byte loads per lane.
__device__ __forceinline__ v16bf load_frag16(const __bf16* __restrict__ base,
                                             int ld, int row, int k0, int lane) {
  const int r  = lane & 15;
  const int kc = (lane >> 4) << 3;
  const __bf16* p = base + (size_t)(row + r) * ld + (k0 + kc);
  Frag16 f;
  f.h[0] = *(const v8bf*)(p);
  f.h[1] = *(const v8bf*)(p + 16);
  return f.v;
}

#define LDSTR 40   // padded LDS row stride (bf16) for 32-elem rows: 80B, bank-conflict-free

__device__ __forceinline__ v16bf lds_frag16(const __bf16* tile0, int lane) {
  const int r  = lane & 15;
  const int kc = (lane >> 4) << 3;
  const __bf16* p = tile0 + r * LDSTR + kc;
  Frag16 f;
  f.h[0] = *(const v8bf*)(p);
  f.h[1] = *(const v8bf*)(p + 16);
  return f.v;
}

// ================= LDS-staged, async double-buffered GEMM =================
// C[M,N] = A[M,K] * W[N,K]^T  (bf16 row-major operands, fp32 out).
// Block: 128 threads = 2x2 waves; block tile 128x128; K step 32.
// A/B 128x32 tiles staged in LDS (double buffered, 20KB) via
// GLOBAL_LOAD_ASYNC_TO_LDS_B128 + s_wait_asynccnt pipelining.
__global__ __launch_bounds__(128)
void gemm_bf16_lds(const __bf16* __restrict__ A, const __bf16* __restrict__ W,
                   float* __restrict__ C, int K, int lda, int ldw, int ldc) {
  __shared__ __align__(16) __bf16 smem[2][2][128 * LDSTR];

  const int tid  = threadIdx.x;
  const int lane = tid & 31;
  const int wave = tid >> 5;
  const int wm   = wave >> 1, wn = wave & 1;
  const int mBlk = blockIdx.y * 128;
  const int nBlk = blockIdx.x * 128;

  // each thread stages one 32-element row of the A tile and of the B tile
  const __bf16* Arow = A + (size_t)(mBlk + tid) * lda;
  const __bf16* Wrow = W + (size_t)(nBlk + tid) * ldw;

  auto issue = [&](int buf, int k0) {
    __bf16* sa = &smem[buf][0][tid * LDSTR];
    __bf16* sb = &smem[buf][1][tid * LDSTR];
#if ASYNC_OK
    #pragma unroll
    for (int c = 0; c < 4; ++c) {
      __builtin_amdgcn_global_load_async_to_lds_b128(
          (GLOBAL_AS v4i*)(Arow + k0 + c * 8), (LDS_AS v4i*)(sa + c * 8), 0, 0);
      __builtin_amdgcn_global_load_async_to_lds_b128(
          (GLOBAL_AS v4i*)(Wrow + k0 + c * 8), (LDS_AS v4i*)(sb + c * 8), 0, 0);
    }
#else
    #pragma unroll
    for (int c = 0; c < 4; ++c) {
      *(v8bf*)(sa + c * 8) = *(const v8bf*)(Arow + k0 + c * 8);
      *(v8bf*)(sb + c * 8) = *(const v8bf*)(Wrow + k0 + c * 8);
    }
#endif
  };

  v8f acc[4][4];
  #pragma unroll
  for (int i = 0; i < 4; ++i)
    #pragma unroll
    for (int j = 0; j < 4; ++j) acc[i][j] = {};

  const int KT = K >> 5;
  issue(0, 0);                                   // prefetch first tile

  for (int kt = 0; kt < KT; ++kt) {
    const int buf = kt & 1;
    if (kt + 1 < KT) { issue(buf ^ 1, (kt + 1) << 5); WAIT_ASYNC(8); }
    else             { WAIT_ASYNC(0); }
    __syncthreads();                             // tiles for `buf` visible

    const __bf16* ta = &smem[buf][0][(wm * 64) * LDSTR];
    const __bf16* tb = &smem[buf][1][(wn * 64) * LDSTR];
    v16bf a[4], b[4];
    #pragma unroll
    for (int i = 0; i < 4; ++i) a[i] = lds_frag16(ta + (i * 16) * LDSTR, lane);
    #pragma unroll
    for (int j = 0; j < 4; ++j) b[j] = lds_frag16(tb + (j * 16) * LDSTR, lane);
    #pragma unroll
    for (int i = 0; i < 4; ++i)
      #pragma unroll
      for (int j = 0; j < 4; ++j)
        acc[i][j] = __builtin_amdgcn_wmma_f32_16x16x32_bf16(
            false, a[i], false, b[j], (short)0, acc[i][j], false, false);

    __syncthreads();                             // done reading `buf`
  }

  // C/D layout: element e <-> row = e + 8*(lane>>4), col = lane&15
  const int rb  = (lane >> 4) << 3;
  const int col = lane & 15;
  const int m0  = mBlk + wm * 64;
  const int n0  = nBlk + wn * 64;
  #pragma unroll
  for (int i = 0; i < 4; ++i)
    #pragma unroll
    for (int j = 0; j < 4; ++j) {
      AccU u; u.v = acc[i][j];
      #pragma unroll
      for (int e = 0; e < 8; ++e)
        C[(size_t)(m0 + i * 16 + rb + e) * ldc + (n0 + j * 16 + col)] = u.f[e];
    }
}

// ============ direct (register-blocked) GEMM for the skinny N=96 ============
template<int WMT, int WNT, int WVM, int WVN>
__global__ __launch_bounds__(32*WVM*WVN)
void gemm_bf16_nt(const __bf16* __restrict__ A, const __bf16* __restrict__ W,
                  float* __restrict__ C, int K, int lda, int ldw, int ldc) {
  const int lane = threadIdx.x & 31;
  const int wave = threadIdx.x >> 5;
  const int m0 = (blockIdx.y * WVM + (wave / WVN)) * (WMT * 16);
  const int n0 = (blockIdx.x * WVN + (wave % WVN)) * (WNT * 16);

  v8f acc[WMT][WNT];
  #pragma unroll
  for (int i = 0; i < WMT; ++i)
    #pragma unroll
    for (int j = 0; j < WNT; ++j) acc[i][j] = {};

  for (int k0 = 0; k0 < K; k0 += 32) {
    v16bf a[WMT], b[WNT];
    #pragma unroll
    for (int i = 0; i < WMT; ++i) a[i] = load_frag16(A, lda, m0 + i*16, k0, lane);
    #pragma unroll
    for (int j = 0; j < WNT; ++j) b[j] = load_frag16(W, ldw, n0 + j*16, k0, lane);
    #pragma unroll
    for (int i = 0; i < WMT; ++i)
      #pragma unroll
      for (int j = 0; j < WNT; ++j)
        acc[i][j] = __builtin_amdgcn_wmma_f32_16x16x32_bf16(
            false, a[i], false, b[j], (short)0, acc[i][j], false, false);
  }

  const int rb  = (lane >> 4) << 3;
  const int col = lane & 15;
  #pragma unroll
  for (int i = 0; i < WMT; ++i)
    #pragma unroll
    for (int j = 0; j < WNT; ++j) {
      AccU u; u.v = acc[i][j];
      #pragma unroll
      for (int e = 0; e < 8; ++e)
        C[(size_t)(m0 + i*16 + rb + e) * ldc + (n0 + j*16 + col)] = u.f[e];
    }
}

// ---------------- elementwise / conv kernels ----------------

__global__ void f32_to_bf16_kernel(const float* __restrict__ s,
                                   __bf16* __restrict__ d, int n) {
  int i = blockIdx.x * blockDim.x + threadIdx.x;
  if (i < n) d[i] = (__bf16)s[i];
}

__global__ void conv_silu_kernel(const float* __restrict__ xz,
                                 const float* __restrict__ cw,
                                 const float* __restrict__ cb,
                                 __bf16* __restrict__ xi) {
  int idx = blockIdx.x * blockDim.x + threadIdx.x;
  if (idx >= MTOT * DI) return;
  int d = idx % DI;
  int t = idx / DI;         // flattened b*LSEQ + l
  int l = t % LSEQ;
  float acc = cb[d];
  #pragma unroll
  for (int j = 0; j < DC; ++j) {
    int ll = l - (DC - 1) + j;
    if (ll >= 0)
      acc = fmaf(cw[d*DC + j], xz[(size_t)(t - (DC - 1) + j) * (2*DI) + d], acc);
  }
  float s = acc / (1.f + __expf(-acc));     // SiLU
  xi[idx] = (__bf16)s;
}

__global__ void dt_softplus_kernel(float* __restrict__ dt,
                                   const float* __restrict__ b_dt) {
  int idx = blockIdx.x * blockDim.x + threadIdx.x;
  if (idx >= MTOT * DI) return;
  float v = dt[idx] + b_dt[idx % DI];
  dt[idx] = (v > 20.f) ? v : log1pf(__expf(v));
}

// ================= chunked selective scan (3 passes) =================
__global__ void scan_pass1(const float* __restrict__ dt,
                           const __bf16* __restrict__ xi,
                           const float* __restrict__ xp,
                           const float* __restrict__ A_log,
                           float* __restrict__ chP, float* __restrict__ chH) {
  int idx = blockIdx.x * blockDim.x + threadIdx.x;
  if (idx >= BSZ * NCHK * DI) return;
  int d = idx % DI;
  int c = (idx / DI) % NCHK;
  int b = idx / (DI * NCHK);

  float Arow[DS], h[DS], P[DS];
  #pragma unroll
  for (int n = 0; n < DS; ++n) {
    Arow[n] = -__expf(A_log[d*DS + n]); h[n] = 0.f; P[n] = 1.f;
  }
  const int l0 = c * CHK;
  for (int l = l0; l < l0 + CHK; ++l) {
    size_t row = (size_t)b * LSEQ + l;
    float dtv = dt[row * DI + d];
    float xv  = (float)xi[row * DI + d];
    const float* xpr = xp + row * XPW;
    float du = dtv * xv;
    #pragma unroll
    for (int n = 0; n < DS; ++n) {
      float e = __expf(dtv * Arow[n]);
      h[n] = e * h[n] + du * xpr[DTR + n];
      P[n] *= e;
    }
  }
  size_t o = (size_t)idx * DS;
  #pragma unroll
  for (int n = 0; n < DS; ++n) { chP[o + n] = P[n]; chH[o + n] = h[n]; }
}

__global__ void scan_pass2(const float* __restrict__ chP,
                           const float* __restrict__ chH,
                           float* __restrict__ chHin) {
  int idx = blockIdx.x * blockDim.x + threadIdx.x;
  if (idx >= BSZ * DI) return;
  int d = idx % DI, b = idx / DI;
  float hin[DS];
  #pragma unroll
  for (int n = 0; n < DS; ++n) hin[n] = 0.f;
  for (int c = 0; c < NCHK; ++c) {
    size_t o = ((size_t)(b * NCHK + c) * DI + d) * DS;
    #pragma unroll
    for (int n = 0; n < DS; ++n) chHin[o + n] = hin[n];
    #pragma unroll
    for (int n = 0; n < DS; ++n) hin[n] = chP[o + n] * hin[n] + chH[o + n];
  }
}

__global__ void scan_pass3(const float* __restrict__ dt,
                           const __bf16* __restrict__ xi,
                           const float* __restrict__ xp,
                           const float* __restrict__ A_log,
                           const float* __restrict__ Dp,
                           const float* __restrict__ xz,
                           const float* __restrict__ chHin,
                           __bf16* __restrict__ yb) {
  int idx = blockIdx.x * blockDim.x + threadIdx.x;
  if (idx >= BSZ * NCHK * DI) return;
  int d = idx % DI;
  int c = (idx / DI) % NCHK;
  int b = idx / (DI * NCHK);

  float Arow[DS], h[DS];
  size_t o = (size_t)idx * DS;
  #pragma unroll
  for (int n = 0; n < DS; ++n) {
    Arow[n] = -__expf(A_log[d*DS + n]);
    h[n]    = chHin[o + n];
  }
  float Dd = Dp[d];
  const int l0 = c * CHK;
  for (int l = l0; l < l0 + CHK; ++l) {
    size_t row = (size_t)b * LSEQ + l;
    float dtv = dt[row * DI + d];
    float xv  = (float)xi[row * DI + d];
    float zv  = xz[row * (2*DI) + DI + d];
    const float* xpr = xp + row * XPW;
    float du = dtv * xv;
    float y  = 0.f;
    #pragma unroll
    for (int n = 0; n < DS; ++n) {
      h[n] = __expf(dtv * Arow[n]) * h[n] + du * xpr[DTR + n];
      y = fmaf(h[n], xpr[DTR + DS + n], y);
    }
    y = fmaf(xv, Dd, y);
    y *= zv / (1.f + __expf(-zv));             // * SiLU(z)
    yb[row * DI + d] = (__bf16)y;
  }
}

// ---------------- host-side launch ----------------
extern "C" void kernel_launch(void* const* d_in, const int* in_sizes, int n_in,
                              void* d_out, int out_size, void* d_ws, size_t ws_size,
                              hipStream_t stream) {
  const float* x      = (const float*)d_in[0];
  const float* W_in   = (const float*)d_in[1];
  const float* conv_w = (const float*)d_in[2];
  const float* conv_b = (const float*)d_in[3];
  const float* W_xp   = (const float*)d_in[4];
  const float* W_dt   = (const float*)d_in[5];
  const float* b_dt   = (const float*)d_in[6];
  const float* A_log  = (const float*)d_in[7];
  const float* Dp     = (const float*)d_in[8];
  const float* W_out  = (const float*)d_in[9];
  float* out = (float*)d_out;

  size_t cur = 0;
  auto alloc = [&](size_t bytes) -> void* {
    char* p = (char*)d_ws + cur;
    cur += (bytes + 255) & ~(size_t)255;
    return (void*)p;
  };
  __bf16* x_b    = (__bf16*)alloc((size_t)MTOT * DM * 2);
  __bf16* Win_b  = (__bf16*)alloc((size_t)2 * DI * DM * 2);
  __bf16* Wxp_b  = (__bf16*)alloc((size_t)XPW * DI * 2);
  __bf16* Wdt_b  = (__bf16*)alloc((size_t)DI * DTR * 2);
  __bf16* Wout_b = (__bf16*)alloc((size_t)DM * DI * 2);
  float*  xz     = (float*) alloc((size_t)MTOT * 2 * DI * 4);
  __bf16* xi_b   = (__bf16*)alloc((size_t)MTOT * DI * 2);
  float*  xp     = (float*) alloc((size_t)MTOT * XPW * 4);
  __bf16* xp_b   = (__bf16*)alloc((size_t)MTOT * XPW * 2);
  float*  dt     = (float*) alloc((size_t)MTOT * DI * 4);
  __bf16* y_b    = (__bf16*)alloc((size_t)MTOT * DI * 2);
  float*  chP    = (float*) alloc((size_t)BSZ * NCHK * DI * DS * 4);
  float*  chH    = (float*) alloc((size_t)BSZ * NCHK * DI * DS * 4);
  float*  chHin  = (float*) alloc((size_t)BSZ * NCHK * DI * DS * 4);

  auto cvt = [&](const float* s, __bf16* d, int n) {
    f32_to_bf16_kernel<<<(n + 255) / 256, 256, 0, stream>>>(s, d, n);
  };
  cvt(x,     x_b,    MTOT * DM);
  cvt(W_in,  Win_b,  2 * DI * DM);
  cvt(W_xp,  Wxp_b,  XPW * DI);
  cvt(W_dt,  Wdt_b,  DI * DTR);
  cvt(W_out, Wout_b, DM * DI);

  // 1) xz = x @ W_in^T            M=16384, N=4096, K=1024
  gemm_bf16_lds<<<dim3((2*DI)/128, MTOT/128), 128, 0, stream>>>(
      x_b, Win_b, xz, DM, DM, DM, 2*DI);

  // 2) xi = SiLU(conv4(xz[:, :DI]))
  conv_silu_kernel<<<(MTOT*DI + 255)/256, 256, 0, stream>>>(xz, conv_w, conv_b, xi_b);

  // 3) xp = xi @ W_xproj^T        M=16384, N=96, K=2048  (128x96 block tile)
  gemm_bf16_nt<2,6,4,1><<<dim3(1, MTOT/128), 128, 0, stream>>>(
      xi_b, Wxp_b, xp, DI, DI, DI, XPW);
  cvt(xp, xp_b, MTOT * XPW);

  // 4) dt_raw = dt_lo @ W_dt^T    M=16384, N=2048, K=64  (dt_lo = xp[:, :64])
  gemm_bf16_lds<<<dim3(DI/128, MTOT/128), 128, 0, stream>>>(
      xp_b, Wdt_b, dt, DTR, XPW, DTR, DI);
  dt_softplus_kernel<<<(MTOT*DI + 255)/256, 256, 0, stream>>>(dt, b_dt);

  // 5) chunked selective scan (+ D skip + SiLU(z) gate) -> y_b (bf16)
  scan_pass1<<<(BSZ*NCHK*DI + 255)/256, 256, 0, stream>>>(
      dt, xi_b, xp, A_log, chP, chH);
  scan_pass2<<<(BSZ*DI + 255)/256, 256, 0, stream>>>(chP, chH, chHin);
  scan_pass3<<<(BSZ*NCHK*DI + 255)/256, 256, 0, stream>>>(
      dt, xi_b, xp, A_log, Dp, xz, chHin, y_b);

  // 6) out = y @ W_out^T          M=16384, N=1024, K=2048
  gemm_bf16_lds<<<dim3(DM/128, MTOT/128), 128, 0, stream>>>(
      y_b, Wout_b, out, DI, DI, DI, DM);
}
